// Reservoir_54692113547479
// MI455X (gfx1250) — compile-verified, matrix operand order checked by reading, not run on hardware
//
#include <hip/hip_runtime.h>
#include <hip/hip_bf16.h>
#include <math.h>

// Reservoir (echo state network) scan for MI455X (gfx1250, wave32).
//   z_s = [r_{s-1} | inp_s] * [A | B]^T + sigma_b      (K fused: 2048 + 96 pad)
//   r_s = (1-alpha) r_{s-1} + alpha tanh(z_s)
// Weights packed once to bf16 (8.8 MB -> L2 resident). Per-step GEMM:
// v_wmma_f32_16x16x32_bf16 with f32 accumulate; 1024 graph-captured step
// kernels; 8-way K split per 16-column tile with LDS reduction.

typedef __attribute__((ext_vector_type(16))) __bf16 v16bf;
typedef __attribute__((ext_vector_type(8)))  float  v8f;

#define NR      2048
#define NU      80
#define NUP     96            // NU padded to multiple of 32
#define KTOT    (NR + NUP)    // 2144 = 67 * 32
#define BATCH   32
#define SEQ     1024
#define ALPHA_F   0.6f
#define SIGMA_B_F 1.6f

// ---------------- prep kernels ----------------

// Pack [A | B | 0-pad] into bf16, row-major (n, k), k-stride KTOT.
__global__ void pack_weights(const float* __restrict__ A,
                             const float* __restrict__ B,
                             __bf16* __restrict__ w) {
    int idx = blockIdx.x * blockDim.x + threadIdx.x;
    int total = NR * KTOT;
    if (idx >= total) return;
    int n = idx / KTOT;
    int k = idx - n * KTOT;
    float v;
    if (k < NR)            v = A[(size_t)n * NR + k];
    else if (k < NR + NU)  v = B[(size_t)n * NU + (k - NR)];
    else                   v = 0.0f;
    w[idx] = (__bf16)v;
}

// Pack input (batch, seq, 80) f32 -> (batch, seq, 96) bf16, zero padded.
__global__ void pack_input(const float* __restrict__ inp,
                           __bf16* __restrict__ xin) {
    int idx = blockIdx.x * blockDim.x + threadIdx.x;
    int total = BATCH * SEQ * NUP;
    if (idx >= total) return;
    int u  = idx % NUP;
    int bs = idx / NUP;
    xin[idx] = (u < NU) ? (__bf16)inp[(size_t)bs * NU + u] : (__bf16)0.0f;
}

// r_0 f32 -> bf16 operand buffer.
__global__ void pack_r0(const float* __restrict__ r0, __bf16* __restrict__ rb) {
    int idx = blockIdx.x * blockDim.x + threadIdx.x;
    if (idx < BATCH * NR) rb[idx] = (__bf16)r0[idx];
}

// ---------------- step kernel ----------------
// One block = one 16-column output tile (gridDim.x = 128), 8 waves.
// r-part K (64 chunks of 32) split 8 per wave, branch-free; the 3 input-part
// chunks are a wave-uniform tail on waves 0..2 (EXEC stays all-1s for WMMA).
// Each wave: 32x16 f32 partial via two v_wmma_f32_16x16x32_bf16 per chunk
// (batch halves share the B operand). 8-way LDS reduce, then tanh + leaky
// update; write f32 trajectory slice and bf16 operand copy for the next step.
__global__ __launch_bounds__(256)
void reservoir_step(const __bf16* __restrict__ w,      // (NR, KTOT) bf16
                    const __bf16* __restrict__ xin,    // (BATCH, SEQ, NUP) bf16
                    const __bf16* __restrict__ rb_in,  // (BATCH, NR) bf16
                    __bf16*       __restrict__ rb_out, // (BATCH, NR) bf16
                    const float*  __restrict__ r0,     // (BATCH, NR) f32
                    float*        __restrict__ out,    // (BATCH, SEQ, NR) f32
                    int s) {
    const int tid  = threadIdx.x;
    const int wave = tid >> 5;         // 0..7
    const int lane = tid & 31;
    const int row  = lane & 15;        // M (batch) row within a half / N column
    const int klo  = (lane >> 4) * 16; // K-half split across lane groups
    const int n0   = blockIdx.x * 16;

    v8f c0 = {}; // batches 0..15  x cols n0..n0+15
    v8f c1 = {}; // batches 16..31 x cols n0..n0+15

    // Recurrent part: 64 chunks, 8 per wave, branch-free constant-trip loop.
    {
        const int kbase = wave * 8 * 32 + klo;
        const __bf16* a0p = rb_in + (size_t)row * NR + kbase;
        const __bf16* a1p = a0p + (size_t)16 * NR;
        const __bf16* bp  = w + (size_t)(n0 + row) * KTOT + kbase;
#pragma unroll 2
        for (int c = 0; c < 8; ++c) {
            v16bf a0 = *(const v16bf*)(a0p + c * 32);
            v16bf a1 = *(const v16bf*)(a1p + c * 32);
            v16bf b  = *(const v16bf*)(bp  + c * 32);
            c0 = __builtin_amdgcn_wmma_f32_16x16x32_bf16(false, a0, false, b,
                                                         (short)0, c0, false, false);
            c1 = __builtin_amdgcn_wmma_f32_16x16x32_bf16(false, a1, false, b,
                                                         (short)0, c1, false, false);
        }
    }

    // Input-projection part: 3 chunks of 32, one each on waves 0..2.
    // Wave-uniform predicate -> EXEC remains all ones inside.
    if (wave < 3) {
        const int ku = wave * 32 + klo;
        v16bf a0 = *(const v16bf*)(xin + ((size_t)row * SEQ + s) * NUP + ku);
        v16bf a1 = *(const v16bf*)(xin + (((size_t)row + 16) * SEQ + s) * NUP + ku);
        v16bf b  = *(const v16bf*)(w + (size_t)(n0 + row) * KTOT + NR + ku);
        c0 = __builtin_amdgcn_wmma_f32_16x16x32_bf16(false, a0, false, b,
                                                     (short)0, c0, false, false);
        c1 = __builtin_amdgcn_wmma_f32_16x16x32_bf16(false, a1, false, b,
                                                     (short)0, c1, false, false);
    }

    // 8-way cross-wave reduction of the 32x16 partials through LDS (16 KB).
    __shared__ float red[8 * 512];
    {
        float* p = red + wave * 512;
#pragma unroll
        for (int i = 0; i < 8; ++i) {
            p[i * 32 + lane]       = c0[i];
            p[256 + i * 32 + lane] = c1[i];
        }
    }
    __syncthreads();

    // 512 output elements, 256 threads -> 2 each.
#pragma unroll
    for (int q = 0; q < 2; ++q) {
        const int e = tid + q * 256;
        float sum = 0.0f;
#pragma unroll
        for (int v = 0; v < 8; ++v) sum += red[v * 512 + e];
        // Accumulator layout: vgpr i, lanes 0-15 -> M=i, lanes 16-31 -> M=i+8.
        const int h = e >> 8;        // batch half
        const int i = (e >> 5) & 7;  // vgpr index
        const int l = e & 31;        // lane
        const int m = h * 16 + ((l < 16) ? i : (i + 8));
        const int n = n0 + (l & 15);
        const float z  = sum + SIGMA_B_F;
        const float rp = (s == 0) ? r0[(size_t)m * NR + n]
                                  : out[((size_t)m * SEQ + (s - 1)) * NR + n];
        const float rn = (1.0f - ALPHA_F) * rp + ALPHA_F * tanhf(z);
        out[((size_t)m * SEQ + s) * NR + n] = rn;
        rb_out[(size_t)m * NR + n] = (__bf16)rn;
    }
}

// r_final = reservoir[:, SEQ-1, :]
__global__ void finalize(const float* __restrict__ out, float* __restrict__ rf) {
    int idx = blockIdx.x * blockDim.x + threadIdx.x;
    if (idx >= BATCH * NR) return;
    int m = idx / NR;
    int n = idx - m * NR;
    rf[idx] = out[((size_t)m * SEQ + (SEQ - 1)) * NR + n];
}

// ---------------- launch ----------------

extern "C" void kernel_launch(void* const* d_in, const int* in_sizes, int n_in,
                              void* d_out, int out_size, void* d_ws, size_t ws_size,
                              hipStream_t stream) {
    const float* inp = (const float*)d_in[0]; // (32, 1024, 80)
    const float* r0  = (const float*)d_in[1]; // (1, 32, 2048)
    const float* A   = (const float*)d_in[2]; // (2048, 2048)
    const float* B   = (const float*)d_in[3]; // (2048, 80)

    float* out = (float*)d_out;                            // (32, 1024, 2048)
    float* rf  = (float*)d_out + (size_t)BATCH * SEQ * NR; // (1, 32, 2048)

    // Workspace layout (bytes):
    //   weights bf16 : NR*KTOT*2        = 8,781,824
    //   input   bf16 : BATCH*SEQ*NUP*2  = 6,291,456
    //   rb[0]   bf16 : BATCH*NR*2       =   131,072
    //   rb[1]   bf16 : BATCH*NR*2       =   131,072
    char* base = (char*)d_ws;
    __bf16* w   = (__bf16*)base;
    __bf16* xin = (__bf16*)(base + (size_t)NR * KTOT * 2);
    __bf16* rbA = (__bf16*)(base + (size_t)NR * KTOT * 2 + (size_t)BATCH * SEQ * NUP * 2);
    __bf16* rbB = rbA + (size_t)BATCH * NR;

    // Prep: pack weights / input / r0 to bf16.
    {
        int total = NR * KTOT;
        pack_weights<<<(total + 255) / 256, 256, 0, stream>>>(A, B, w);
    }
    {
        int total = BATCH * SEQ * NUP;
        pack_input<<<(total + 255) / 256, 256, 0, stream>>>(inp, xin);
    }
    {
        int total = BATCH * NR;
        pack_r0<<<(total + 255) / 256, 256, 0, stream>>>(r0, rbA);
    }

    // Sequential scan: 1024 step kernels, ping-ponging the bf16 state operand.
    for (int s = 0; s < SEQ; ++s) {
        __bf16* rin  = (s & 1) ? rbB : rbA;
        __bf16* rout = (s & 1) ? rbA : rbB;
        reservoir_step<<<NR / 16, 256, 0, stream>>>(w, xin, rin, rout, r0, out, s);
    }

    finalize<<<(BATCH * NR + 255) / 256, 256, 0, stream>>>(out, rf);
}
